// SpectralAttention_71150428225652
// MI455X (gfx1250) — compile-verified, matrix operand order checked by reading, not run on hardware
//
#include <hip/hip_runtime.h>
#include <hip/hip_bf16.h>

// SpectralAttention for MI455X (gfx1250, wave32, WMMA + TDM/async-LDS).
// Pipeline (all on `stream`):
//  1. cast fp32 -> bf16 (X, Wq, Wk, Wv, Wo)
//  2. wmma_gemm: Q (1/sqrt(HD) folded), K, V -> [B,H,S,HD] bf16
//     (LDS double-buffered; A tile via TDM tensor_load_to_lds, B tile via
//      global_load_async_to_lds_b128; fragments fed from LDS)
//  3. wmma_gemm batched: scores = Q K^T (fp32)
//  4. fft_rows: real->complex row FFT (radix-2 DIT in LDS)
//  5. fft_cols_filter: col FFT + exp(i*alpha*atan(log|F|+eps)) + col IFFT fused;
//     16x1024 complex tile = 128 KB LDS, loaded via async-to-LDS b64 scatter
//  6. ifft_rows_softmax: row IFFT + real + softmax fused -> attn bf16
//  7. ctx_gemm: attn @ V; whole 128 KB V panel staged by ONE TDM descriptor
//  8. wmma_gemm: out = ctx @ Wo^T + bo (fp32)

#define DEVINL __device__ __forceinline__

typedef __attribute__((ext_vector_type(16))) __bf16        v16bf;
typedef __attribute__((ext_vector_type(8)))  float         v8f;
typedef __attribute__((ext_vector_type(4)))  unsigned int  v4u;
typedef __attribute__((ext_vector_type(8)))  int           v8i;
typedef __attribute__((ext_vector_type(4)))  int           v4i;
// gcc-style vectors: exact pointee types of the async-to-LDS builtins
typedef int gvi4 __attribute__((vector_size(16)));
typedef int gvi2 __attribute__((vector_size(8)));

union FragBF { v4u q[2]; v16bf v; unsigned short s[16]; };

static_assert(sizeof(v16bf) == 32, "v16bf must be 32 bytes");

// ------------------------------------------------ CDNA5 feature detection
#if defined(__HIP_DEVICE_COMPILE__) && defined(__gfx1250__)
# if __has_builtin(__builtin_amdgcn_global_load_async_to_lds_b128) && \
     __has_builtin(__builtin_amdgcn_global_load_async_to_lds_b64)  && \
     __has_builtin(__builtin_amdgcn_s_wait_asynccnt)
#  define HAVE_ASYNC_LDS 1
# endif
# if __has_builtin(__builtin_amdgcn_tensor_load_to_lds) && \
     __has_builtin(__builtin_amdgcn_s_wait_tensorcnt)
#  define HAVE_TDM 1
# endif
#endif
#ifndef HAVE_ASYNC_LDS
#define HAVE_ASYNC_LDS 0
#endif
#ifndef HAVE_TDM
#define HAVE_TDM 0
#endif

// ------------------------------------------------ helpers

DEVINL unsigned short f2bf(float f) {
  unsigned int u = __float_as_uint(f);
  u += 0x7FFFu + ((u >> 16) & 1u);   // round-to-nearest-even
  return (unsigned short)(u >> 16);
}

// 16-byte global->LDS copy; async (ASYNCcnt) when available.
DEVINL void async_copy_b128(void* lds_dst, const void* gsrc) {
#if HAVE_ASYNC_LDS
  __builtin_amdgcn_global_load_async_to_lds_b128(
      (__attribute__((address_space(1))) gvi4*)(unsigned long long)gsrc,
      (__attribute__((address_space(3))) gvi4*)lds_dst, 0, 0);
#else
  *(v4u*)lds_dst = *(const v4u*)gsrc;
#endif
}

// 8-byte global->LDS copy (per-lane LDS scatter address is allowed).
DEVINL void async_copy_b64(void* lds_dst, const void* gsrc) {
#if HAVE_ASYNC_LDS
  __builtin_amdgcn_global_load_async_to_lds_b64(
      (__attribute__((address_space(1))) gvi2*)(unsigned long long)gsrc,
      (__attribute__((address_space(3))) gvi2*)lds_dst, 0, 0);
#else
  *(float2*)lds_dst = *(const float2*)gsrc;
#endif
}

#if HAVE_TDM
// LDS byte offset of a generic pointer known to point into LDS.
DEVINL unsigned lds_addr_of(const void* p) {
  return (unsigned)(unsigned long long)
         (__attribute__((address_space(3))) const void*)p;
}

// Issue one TDM descriptor; the builtin is 5-arg on ROCm clang-22 and 6-arg
// (extra zero int32x8 = unused VADDR4 group) on upstream clang-23.
DEVINL void tdm_issue(v4u g0, v8i g1) {
  const v4i zz = {0, 0, 0, 0};
#if __clang_major__ <= 22
  __builtin_amdgcn_tensor_load_to_lds(g0, g1, zz, zz, 0);
#else
  const v8i zz8 = {0, 0, 0, 0, 0, 0, 0, 0};
  __builtin_amdgcn_tensor_load_to_lds(g0, g1, zz, zz, zz8, 0);
#endif
}

// One TDM descriptor: load a tight 2-D tile (tile_w elems x tile_h rows of
// 2-byte data, row pitch row_stride elems) from global to LDS (packed rows).
// D# layout per CDNA5 ISA 8.3/8.4: group0 = {count=1, lds_addr, global_addr,
// type=2}; group1 = {data_size=1(2B), tensor_dim0/1 = tile dims (tight),
// tile_dim0/1, tensor_dim0_stride}; groups 2/3 zero (<=2-D tensor).
DEVINL void tdm_load_2d(unsigned lds_byte_addr, const void* gaddr,
                        unsigned tile_w, unsigned tile_h,
                        unsigned long long row_stride) {
  const unsigned long long ga = (unsigned long long)gaddr;
  v4u g0;
  g0.x = 1u;                                             // count=1, user load
  g0.y = lds_byte_addr;                                  // lds_addr
  g0.z = (unsigned)ga;                                   // global_addr lo
  g0.w = (unsigned)((ga >> 32) & 0x01FFFFFFull) | (2u << 30);  // hi25 | type=2
  v8i g1;
  g1[0] = (int)(1u << 16);                               // data_size = 2 bytes
  g1[1] = (int)((tile_w & 0xFFFFu) << 16);               // tensor_dim0 lo16
  g1[2] = (int)(((tile_w >> 16) & 0xFFFFu) |             // tensor_dim0 hi16
                ((tile_h & 0xFFFFu) << 16));             // tensor_dim1 lo16
  g1[3] = (int)(((tile_h >> 16) & 0xFFFFu) |             // tensor_dim1 hi16
                ((tile_w & 0xFFFFu) << 16));             // tile_dim0
  g1[4] = (int)(tile_h & 0xFFFFu);                       // tile_dim1 (dim2=0)
  g1[5] = (int)(row_stride & 0xFFFFFFFFull);             // dim0_stride lo32
  g1[6] = (int)((row_stride >> 32) & 0xFFFFull);         // dim0_stride hi16
  g1[7] = 0;
  tdm_issue(g0, g1);
}
#endif

// Load one 16x32 bf16 fragment (A layout per CDNA5 ISA 7.12.2): lanes 0-15 hold
// row M=lane with k-chunks {0..7,16..23}; lanes 16-31 row M=lane-16 with
// {8..15,24..31}. B fragments use the mirrored layout (lane = N); all our B
// sources here are N-major x K-contiguous, so the same loader works.
DEVINL v16bf load_frag_k32(const unsigned short* tile, int row_stride, int lane) {
  const int r  = lane & 15;
  const int ko = (lane & 16) ? 8 : 0;
  const unsigned short* p = tile + (size_t)r * (size_t)row_stride + ko;
  FragBF f;
  f.q[0] = *(const v4u*)(p);
  f.q[1] = *(const v4u*)(p + 16);
  return f.v;
}

// ------------------------------------------------ cast

__global__ void cast_f32_bf16_kernel(const float* __restrict__ src,
                                     unsigned short* __restrict__ dst, int n) {
  int i = blockIdx.x * blockDim.x + threadIdx.x;
  if (i < n) dst[i] = f2bf(src[i]);
}

// ------------------------------------------------ WMMA GEMM (LDS dbl-buffered)
// C[m,n] = sum_k A[m,k] * B[n,k] (row-major, K-contiguous rows).
// 256 threads = 8 waves; block tile 128x64, k-step 32.
// A tile 128x32 staged by TDM (wave 0), B tile 64x32 by async-to-LDS b128.
// mode 0: bf16 output permuted to [B,H,S,HD]; mode 1: fp32 output.

__global__ void wmma_gemm_kernel(const unsigned short* __restrict__ A, long long strideA,
                                 const unsigned short* __restrict__ Bm, long long strideB,
                                 const float* __restrict__ bias,
                                 unsigned short* __restrict__ out_bf,
                                 float* __restrict__ out_f32, long long strideO,
                                 int K, int ldOut, int mode, float scale) {
  __shared__ __align__(16) unsigned short Asm[2][128 * 32];   // 2 x 8 KB
  __shared__ __align__(16) unsigned short Bsm[2][64 * 32];    // 2 x 4 KB

  const int tid  = threadIdx.x;
  const int lane = tid & 31;
  const int wave = tid >> 5;
  const int m0 = blockIdx.y * 128 + wave * 16;
  const int n0 = blockIdx.x * 64;

  const unsigned short* Ablk =
      A + (size_t)blockIdx.z * (size_t)strideA + (size_t)(blockIdx.y * 128) * (size_t)K;
  const unsigned short* Bblk =
      Bm + (size_t)blockIdx.z * (size_t)strideB + (size_t)n0 * (size_t)K;

#if HAVE_TDM
  const unsigned asm_lds = lds_addr_of(&Asm[0][0]);
#endif

  auto stage_tile = [&](int buf, int kb) {
#if HAVE_TDM
    if (wave == 0)
      tdm_load_2d(asm_lds + (unsigned)buf * (128 * 32 * 2), Ablk + kb,
                  32u, 128u, (unsigned long long)K);
#else
    for (int c = tid; c < 512; c += 256) {          // 128x32 = 512 b128 chunks
      const int row = c >> 2, kc = (c & 3) << 3;
      async_copy_b128(&Asm[buf][row * 32 + kc], Ablk + (size_t)row * K + kb + kc);
    }
#endif
    {                                               // 64x32 = 256 b128 chunks
      const int row = tid >> 2, kc = (tid & 3) << 3;
      async_copy_b128(&Bsm[buf][row * 32 + kc], Bblk + (size_t)row * K + kb + kc);
    }
  };

  auto wait_tile = [&](bool pending) {              // wait imms must be literal
#if HAVE_TDM
    if (wave == 0) {
      if (pending) __builtin_amdgcn_s_wait_tensorcnt(1);
      else         __builtin_amdgcn_s_wait_tensorcnt(0);
    }
#endif
#if HAVE_ASYNC_LDS
# if HAVE_TDM
    if (pending) __builtin_amdgcn_s_wait_asynccnt(1);
    else         __builtin_amdgcn_s_wait_asynccnt(0);
# else
    if (pending) __builtin_amdgcn_s_wait_asynccnt(3);
    else         __builtin_amdgcn_s_wait_asynccnt(0);
# endif
#endif
    __syncthreads();
  };

  v8f acc[4] = {};
  stage_tile(0, 0);
  int cur = 0;
  for (int kb = 0; kb < K; kb += 32) {
    const bool has_next = (kb + 32) < K;
    if (has_next) stage_tile(cur ^ 1, kb + 32);
    wait_tile(has_next);

    v16bf af = load_frag_k32(&Asm[cur][wave * 512], 32, lane);
    v16bf bfr[4];
#pragma unroll
    for (int j = 0; j < 4; ++j)
      bfr[j] = load_frag_k32(&Bsm[cur][j * 512], 32, lane);
#pragma unroll
    for (int j = 0; j < 4; ++j)
      acc[j] = __builtin_amdgcn_wmma_f32_16x16x32_bf16(
          false, af, false, bfr[j], (short)0, acc[j], false, false);

    __syncthreads();                                // buffer free for re-stage
    cur ^= 1;
  }

  // C/D layout: lanes 0-15 -> N=lane, rows m0+r; lanes 16-31 -> N=lane-16, m0+8+r
  const int nlane = lane & 15;
  const int mhalf = (lane & 16) ? 8 : 0;
#pragma unroll
  for (int j = 0; j < 4; ++j) {
    const int n = n0 + j * 16 + nlane;
    const float bs = bias ? bias[n] : 0.f;
#pragma unroll
    for (int r = 0; r < 8; ++r) {
      const int m = m0 + mhalf + r;
      const float val = (acc[j][r] + bs) * scale;
      if (mode == 0) {
        const int b = m >> 10, s = m & 1023;
        const int h = n >> 6,  d = n & 63;
        const size_t idx = ((size_t)(b * 16 + h) << 16) + ((size_t)s << 6) + (size_t)d;
        out_bf[idx] = f2bf(val);
      } else {
        out_f32[(size_t)blockIdx.z * (size_t)strideO + (size_t)m * (size_t)ldOut + n] = val;
      }
    }
  }
}

// ------------------------------------------------ FFT core (radix-2 DIT)

DEVINL void cmul_butterfly(float2& a, float2& b, float c, float s) {
  const float tbx = b.x * c - b.y * s;
  const float tby = b.x * s + b.y * c;
  b.x = a.x - tbx; b.y = a.y - tby;
  a.x += tbx;      a.y += tby;
}

DEVINL void fft1024_stages(float2* lds, int tid, int ncols) {
  for (int half = 1; half < 1024; half <<= 1) {
    const float astep = -3.14159265358979f / (float)half;
    for (int t = tid; t < ncols * 512; t += 256) {
      const int col = t >> 9;
      const int bt  = t & 511;
      const int j   = bt & (half - 1);
      const int idx = ((bt & ~(half - 1)) << 1) + j;
      float sn, cs;
      __sincosf(astep * (float)j, &sn, &cs);
      float2* cb = lds + col * 1024;
      cmul_butterfly(cb[idx], cb[idx + half], cs, sn);
    }
    __syncthreads();
  }
}

DEVINL int brev10(int i) { return (int)(__brev((unsigned)i) >> 22); }

// Pass A: forward FFT along rows (axis k), real input -> complex output.
__global__ void fft_rows_kernel(const float* __restrict__ scores,
                                float2* __restrict__ cplx) {
  __shared__ float2 buf[1024];
  const int tid = threadIdx.x;
  const size_t rb = (size_t)blockIdx.x * 1024;
  for (int i = tid; i < 1024; i += 256)
    buf[brev10(i)] = make_float2(scores[rb + i], 0.f);
  __syncthreads();
  fft1024_stages(buf, tid, 1);
  for (int i = tid; i < 1024; i += 256) cplx[rb + i] = buf[i];
}

// Pass B: column FFT + spectral filter + column IFFT, fused.
// 16 columns x 1024 rows = 128 KB dynamic LDS (CDNA5: 320 KB/WGP).
__global__ void fft_cols_filter_kernel(float2* __restrict__ cplx,
                                       const float* __restrict__ alpha_p) {
  extern __shared__ float2 lds[];  // [16][1024] column-major tiles
  const int tid = threadIdx.x;
  const int c0  = blockIdx.x * 16;
  float2* base = cplx + (size_t)blockIdx.y * (1024u * 1024u);
  const float alpha = *alpha_p;

  // async global->LDS, per-lane bit-reversed LDS scatter addresses
  for (int i = tid; i < 16 * 1024; i += 256) {
    const int row = i >> 4, c = i & 15;
    async_copy_b64(&lds[c * 1024 + brev10(row)], base + (size_t)row * 1024 + c0 + c);
  }
#if HAVE_ASYNC_LDS
  __builtin_amdgcn_s_wait_asynccnt(0);
#endif
  __syncthreads();

  fft1024_stages(lds, tid, 16);   // forward FFT along columns

  // z *= exp(i*alpha*atan(log|z|+eps)); conj to set up inverse
  for (int i = tid; i < 16 * 1024; i += 256) {
    float2 z = lds[i];
    const float mag = sqrtf(z.x * z.x + z.y * z.y);
    const float ph  = alpha * atanf(__logf(mag + 1e-10f));
    float sn, cs;
    __sincosf(ph, &sn, &cs);
    float2 r;
    r.x = z.x * cs - z.y * sn;
    r.y = -(z.x * sn + z.y * cs);
    lds[i] = r;
  }
  __syncthreads();

  // in-place per-column bit-reversal permutation
  for (int i = tid; i < 16 * 1024; i += 256) {
    const int row = i & 1023, c = i >> 10;
    const int br = brev10(row);
    if (row < br) {
      float2 t0 = lds[c * 1024 + row];
      lds[c * 1024 + row] = lds[c * 1024 + br];
      lds[c * 1024 + br]  = t0;
    }
  }
  __syncthreads();

  fft1024_stages(lds, tid, 16);   // IFFT = conj(FFT(conj(z)))/N

  for (int i = tid; i < 16 * 1024; i += 256) {
    const int row = i >> 4, c = i & 15;
    float2 z = lds[c * 1024 + row];
    z.x *=  (1.0f / 1024.0f);
    z.y *= -(1.0f / 1024.0f);
    base[(size_t)row * 1024 + c0 + c] = z;
  }
}

// Pass C: row IFFT + real part + softmax over keys, fused. Emits bf16 attn.
__global__ void ifft_rows_softmax_kernel(const float2* __restrict__ cplx,
                                         unsigned short* __restrict__ attn) {
  __shared__ float2 buf[1024];
  __shared__ float  red[256];
  const int tid = threadIdx.x;
  const size_t rb = (size_t)blockIdx.x * 1024;

  for (int i = tid; i < 1024; i += 256) {
    float2 z = cplx[rb + i];
    z.y = -z.y;
    buf[brev10(i)] = z;
  }
  __syncthreads();
  fft1024_stages(buf, tid, 1);

  float lmax = -3.4e38f;
  for (int i = tid; i < 1024; i += 256) {
    const float x = buf[i].x * (1.0f / 1024.0f);
    buf[i].x = x;
    lmax = fmaxf(lmax, x);
  }
  red[tid] = lmax;
  __syncthreads();
  for (int s2 = 128; s2 > 0; s2 >>= 1) {
    if (tid < s2) red[tid] = fmaxf(red[tid], red[tid + s2]);
    __syncthreads();
  }
  const float mx = red[0];
  __syncthreads();

  float lsum = 0.f;
  for (int i = tid; i < 1024; i += 256) {
    const float e = __expf(buf[i].x - mx);
    buf[i].x = e;
    lsum += e;
  }
  red[tid] = lsum;
  __syncthreads();
  for (int s2 = 128; s2 > 0; s2 >>= 1) {
    if (tid < s2) red[tid] += red[tid + s2];
    __syncthreads();
  }
  const float inv = 1.0f / red[0];
  for (int i = tid; i < 1024; i += 256)
    attn[rb + i] = f2bf(buf[i].x * inv);
}

// ------------------------------------------------ ctx = attn @ V
// Whole per-(b,h) V panel [1024 k][64 d] bf16 = 128 KB staged to LDS by one
// TDM descriptor (async b128 / plain-copy fallback). B fragments gathered from
// LDS (k-strided, per the B-matrix lane=N layout); A streamed from global.
__global__ void ctx_gemm_kernel(const unsigned short* __restrict__ attn,
                                const unsigned short* __restrict__ Vbf,
                                unsigned short* __restrict__ ctx) {
  extern __shared__ unsigned short Vlds[];   // [1024][64]
  const int tid  = threadIdx.x;
  const int lane = tid & 31;
  const int wave = tid >> 5;
  const int bh   = blockIdx.z;               // b*16 + h

  const unsigned short* Vb = Vbf + (size_t)bh * 65536;
#if HAVE_TDM
  if (wave == 0) tdm_load_2d(lds_addr_of(Vlds), Vb, 64u, 1024u, 64ull);
  if (wave == 0) __builtin_amdgcn_s_wait_tensorcnt(0);
#else
  for (int c = tid; c < 8192; c += 256)      // 8192 b128 chunks
    async_copy_b128(&Vlds[c * 8], Vb + (size_t)c * 8);
# if HAVE_ASYNC_LDS
  __builtin_amdgcn_s_wait_asynccnt(0);
# endif
#endif
  __syncthreads();

  const int m0 = blockIdx.y * 128 + wave * 16;   // query rows
  const unsigned short* Ab =
      attn + (size_t)bh * (1024u * 1024u) + (size_t)m0 * 1024;

  const int nl = lane & 15;
  const int ko = (lane & 16) ? 8 : 0;

  v8f acc[4] = {};
  for (int kb = 0; kb < 1024; kb += 32) {
    __builtin_prefetch(Ab + kb + 256, 0, 1);      // global_prefetch_b8
    v16bf af = load_frag_k32(Ab + kb, 1024, lane);
    v16bf bfr[4];
#pragma unroll
    for (int j = 0; j < 4; ++j) {
      const int n = j * 16 + nl;
      FragBF f;
#pragma unroll
      for (int t = 0; t < 8; ++t) {
        f.s[t]     = Vlds[(size_t)(kb + ko + t) * 64 + n];
        f.s[8 + t] = Vlds[(size_t)(kb + 16 + ko + t) * 64 + n];
      }
      bfr[j] = f.v;
    }
#pragma unroll
    for (int j = 0; j < 4; ++j)
      acc[j] = __builtin_amdgcn_wmma_f32_16x16x32_bf16(
          false, af, false, bfr[j], (short)0, acc[j], false, false);
  }

  // store ctx[b, s, h*64 + d] bf16
  const int b = bh >> 4, h = bh & 15;
  const int mhalf = (lane & 16) ? 8 : 0;
#pragma unroll
  for (int j = 0; j < 4; ++j) {
    const int d = j * 16 + nl;
#pragma unroll
    for (int r = 0; r < 8; ++r) {
      const int s = m0 + mhalf + r;
      const size_t idx = ((size_t)b * 1024 + s) * 1024 + (size_t)h * 64 + d;
      ctx[idx] = f2bf(acc[j][r]);
    }
  }
}

// ------------------------------------------------ launcher

extern "C" void kernel_launch(void* const* d_in, const int* in_sizes, int n_in,
                              void* d_out, int out_size, void* d_ws, size_t ws_size,
                              hipStream_t stream) {
  (void)in_sizes; (void)n_in; (void)out_size; (void)ws_size;

  const float* query = (const float*)d_in[0];
  const float* Wq    = (const float*)d_in[1];
  const float* bq    = (const float*)d_in[2];
  const float* Wk    = (const float*)d_in[3];
  const float* bk    = (const float*)d_in[4];
  const float* Wv    = (const float*)d_in[5];
  const float* bv    = (const float*)d_in[6];
  const float* Wo    = (const float*)d_in[7];
  const float* bo    = (const float*)d_in[8];
  const float* alpha = (const float*)d_in[9];

  char* ws = (char*)d_ws;
  constexpr size_t MB = 1024ull * 1024ull;
  unsigned short* Xbf   = (unsigned short*)(ws + 0 * MB);    //  8 MB
  unsigned short* Wqbf  = (unsigned short*)(ws + 8 * MB);    //  2 MB
  unsigned short* Wkbf  = (unsigned short*)(ws + 10 * MB);   //  2 MB
  unsigned short* Wvbf  = (unsigned short*)(ws + 12 * MB);   //  2 MB
  unsigned short* Wobf  = (unsigned short*)(ws + 14 * MB);   //  2 MB
  unsigned short* Qbf   = (unsigned short*)(ws + 16 * MB);   //  8 MB [B,H,S,HD]
  unsigned short* Kbf   = (unsigned short*)(ws + 24 * MB);   //  8 MB
  unsigned short* Vbf   = (unsigned short*)(ws + 32 * MB);   //  8 MB
  unsigned short* ctxbf = (unsigned short*)(ws + 40 * MB);   //  8 MB [B,S,E]
  float*          score = (float*)(ws + 48 * MB);            // 256 MB
  unsigned short* attn  = (unsigned short*)(ws + 48 * MB);   // 128 MB (aliases score)
  float2*         cplx  = (float2*)(ws + 304 * MB);          // 512 MB

  // 1. casts
  cast_f32_bf16_kernel<<<(4194304 + 255) / 256, 256, 0, stream>>>(query, Xbf, 4194304);
  cast_f32_bf16_kernel<<<(1048576 + 255) / 256, 256, 0, stream>>>(Wq, Wqbf, 1048576);
  cast_f32_bf16_kernel<<<(1048576 + 255) / 256, 256, 0, stream>>>(Wk, Wkbf, 1048576);
  cast_f32_bf16_kernel<<<(1048576 + 255) / 256, 256, 0, stream>>>(Wv, Wvbf, 1048576);
  cast_f32_bf16_kernel<<<(1048576 + 255) / 256, 256, 0, stream>>>(Wo, Wobf, 1048576);

  // 2. Q/K/V projections (M=4096, N=1024, K=1024)
  const dim3 gProj(16, 32, 1);
  wmma_gemm_kernel<<<gProj, 256, 0, stream>>>(Xbf, 0, Wqbf, 0, bq, Qbf, nullptr, 0,
                                              1024, 1024, 0, 0.125f);
  wmma_gemm_kernel<<<gProj, 256, 0, stream>>>(Xbf, 0, Wkbf, 0, bk, Kbf, nullptr, 0,
                                              1024, 1024, 0, 1.0f);
  wmma_gemm_kernel<<<gProj, 256, 0, stream>>>(Xbf, 0, Wvbf, 0, bv, Vbf, nullptr, 0,
                                              1024, 1024, 0, 1.0f);

  // 3. scores = Q K^T, batched over 64 (b,h) planes (M=N=1024, K=64)
  const dim3 gScore(16, 8, 64);
  wmma_gemm_kernel<<<gScore, 256, 0, stream>>>(Qbf, 65536, Kbf, 65536, nullptr,
                                               nullptr, score, 1048576,
                                               64, 1024, 1, 1.0f);

  // 4. row FFT (real -> complex)
  fft_rows_kernel<<<65536, 256, 0, stream>>>(score, cplx);

  // 5. column FFT + filter + column IFFT (fused), 128 KB dynamic LDS
  fft_cols_filter_kernel<<<dim3(64, 64), 256, 131072, stream>>>(cplx, alpha);

  // 6. row IFFT + real + softmax -> bf16 attn
  ifft_rows_softmax_kernel<<<65536, 256, 0, stream>>>(cplx, attn);

  // 7. ctx = attn @ V, 128 KB V panel staged by one TDM descriptor
  ctx_gemm_kernel<<<dim3(1, 8, 64), 256, 131072, stream>>>(attn, Vbf, ctxbf);

  // 8. out = ctx @ Wo^T + bo (fp32)
  wmma_gemm_kernel<<<gProj, 256, 0, stream>>>(ctxbf, 0, Wobf, 0, bo,
                                              nullptr, (float*)d_out, 0,
                                              1024, 1024, 1, 1.0f);
}
